// EuclideanCodebook_48232482734185
// MI455X (gfx1250) — matrix operand order repaired and checked by reading, not compiled, not run on hardware
//
#include <hip/hip_runtime.h>

typedef __attribute__((ext_vector_type(16))) __bf16 v16bf;
typedef __attribute__((ext_vector_type(8)))  __bf16 v8bf;
typedef __attribute__((ext_vector_type(8)))  float  v8f;
typedef __attribute__((ext_vector_type(4)))  float  v4f;

#define D_DIM 256
#define ROW_BYTES   512                         // 256 bf16 per codebook row
#define ROW_STRIDE  528                         // padded LDS stride: 132 dwords == 4 mod 64 banks
#define STAGE_ROWS  32                          // two 16-code N tiles per stage
#define STAGE_BYTES (STAGE_ROWS * ROW_STRIDE)   // 16896 B per buffer
#define NWAVES      16                          // waves per block
#define NTHREADS    (NWAVES * 32)               // 512

// Low 32 bits of a generic pointer to LDS == LDS byte offset (aperture lives in [63:32]).
__device__ __forceinline__ unsigned lds_off(const void* p) {
  return (unsigned)(unsigned long long)p;
}

// CDNA5 async copy: 16B global -> LDS per lane, tracked by ASYNCcnt.
__device__ __forceinline__ void async_b128(unsigned dst_lds, const void* src) {
  asm volatile("global_load_async_to_lds_b128 %0, %1, off"
               :: "v"(dst_lds), "v"(src) : "memory");
}

// ---------------------------------------------------------------------------
// Prep: embed fp32 -> bf16 (WMMA feed) + e2[c] = sum(bf16(e)^2), fp32.
// One wave per codebook row; 8 floats per lane, coalesced.
// ---------------------------------------------------------------------------
__global__ __launch_bounds__(256) void vq_prep_kernel(const float* __restrict__ embed,
                                                      __bf16* __restrict__ ebf,
                                                      float* __restrict__ e2) {
  const int lane = threadIdx.x & 31;
  const int wave = threadIdx.x >> 5;
  const int c = blockIdx.x * 8 + wave;
  const float* row = embed + (size_t)c * D_DIM;
  const int d0 = lane * 8;

  v4f f0 = *(const v4f*)(row + d0);
  v4f f1 = *(const v4f*)(row + d0 + 4);
  v8bf b;
  b[0] = (__bf16)f0.x; b[1] = (__bf16)f0.y; b[2] = (__bf16)f0.z; b[3] = (__bf16)f0.w;
  b[4] = (__bf16)f1.x; b[5] = (__bf16)f1.y; b[6] = (__bf16)f1.z; b[7] = (__bf16)f1.w;
  *(v8bf*)(ebf + (size_t)c * D_DIM + d0) = b;

  float s = 0.f;
#pragma unroll
  for (int i = 0; i < 8; ++i) { float v = (float)b[i]; s += v * v; }
#pragma unroll
  for (int off = 16; off > 0; off >>= 1) s += __shfl_xor(s, off, 32);
  if (lane == 0) e2[c] = s;
}

// ---------------------------------------------------------------------------
// Fused GEMM + argmax + gather.
// Block = 16 waves x 16 rows = 256 x-rows. Codebook streamed through LDS in
// double-buffered async stages of 32 codes; all 16 waves share each stage
// (halves L2 codebook traffic vs 8-wave blocks).
// score(m,c) = 2 * dot(x_m, e_c) - ||e_c||^2  (argmax == nearest code)
// ---------------------------------------------------------------------------
__global__ __launch_bounds__(NTHREADS) void vq_argmin_kernel(
    const float* __restrict__ x, const float* __restrict__ embed,
    const __bf16* __restrict__ ebf, const float* __restrict__ e2,
    float* __restrict__ quant, float* __restrict__ ind_out, int C) {
  __shared__ __align__(16) char smem[2 * STAGE_BYTES];   // 33792 B

  const int tid  = threadIdx.x;
  const int lane = tid & 31;
  const int wave = tid >> 5;
  const int half = lane >> 4;      // 16-lane group
  const int mcol = lane & 15;      // A: row id / B,C,D: column id
  const int row0 = (blockIdx.x * NWAVES + wave) * 16;

  // ---- A panel (16 rows x 256 K), fp32 -> bf16, held in registers for the
  // whole codebook sweep. WMMA A layout: lanes 0-15 row m=lane carry
  // K=kb..kb+7 (elems 0-7) and K=kb+16..kb+23 (elems 8-15); lanes 16-31
  // same rows with kb shifted by 8.
  v16bf A[8];
  const float* xrow = x + (size_t)(row0 + mcol) * D_DIM;
#pragma unroll
  for (int kk = 0; kk < 8; ++kk) {
    const int kb = kk * 32 + half * 8;
    v4f f0 = *(const v4f*)(xrow + kb);
    v4f f1 = *(const v4f*)(xrow + kb + 4);
    v4f f2 = *(const v4f*)(xrow + kb + 16);
    v4f f3 = *(const v4f*)(xrow + kb + 20);
    v16bf a;
    a[0]  = (__bf16)f0.x; a[1]  = (__bf16)f0.y; a[2]  = (__bf16)f0.z; a[3]  = (__bf16)f0.w;
    a[4]  = (__bf16)f1.x; a[5]  = (__bf16)f1.y; a[6]  = (__bf16)f1.z; a[7]  = (__bf16)f1.w;
    a[8]  = (__bf16)f2.x; a[9]  = (__bf16)f2.y; a[10] = (__bf16)f2.z; a[11] = (__bf16)f2.w;
    a[12] = (__bf16)f3.x; a[13] = (__bf16)f3.y; a[14] = (__bf16)f3.z; a[15] = (__bf16)f3.w;
    A[kk] = a;
  }

  float best[8];
  int   bidx[8];
#pragma unroll
  for (int v = 0; v < 8; ++v) { best[v] = -3.4e38f; bidx[v] = 0; }

  const int ntiles  = C >> 4;        // 128
  const int nstages = ntiles >> 1;   // 64

  // Stage issue: 16 KB of codebook (32 rows x 512 B) -> padded LDS buffer.
  // 512 threads x 1 chunk x 2 async b128 = 2 ASYNCcnt ops per wave.
  auto stage = [&](int s) {
    const char* gbase = (const char*)ebf + (size_t)s * STAGE_ROWS * ROW_BYTES;
    const unsigned lbase = lds_off(smem) + (unsigned)(s & 1) * STAGE_BYTES;
    const int chunk = tid;                 // 512 chunks of 32 B
    const int r   = chunk >> 4;
    const int cch = chunk & 15;
    const unsigned dst = lbase + (unsigned)(r * ROW_STRIDE + cch * 32);
    const char* src = gbase + r * ROW_BYTES + cch * 32;
    async_b128(dst,      src);
    async_b128(dst + 16, src + 16);
  };

  stage(0);
  for (int s = 0; s < nstages; ++s) {
    if (s + 1 < nstages) {
      stage(s + 1);                                     // overlap next stage
      asm volatile("s_wait_asynccnt 0x2" ::: "memory"); // 2 older ops done
    } else {
      asm volatile("s_wait_asynccnt 0x0" ::: "memory");
    }
    __syncthreads();

    const char* buf = smem + (s & 1) * STAGE_BYTES;
#pragma unroll
    for (int t = 0; t < 2; ++t) {
      const int ct = s * 2 + t;
      // B 32x16 layout: lane = column n (code), lanes 0-15 K=kk*32+0..7,+16..23,
      // lanes 16-31 K=kk*32+8..15,+24..31 (matches A).
      const char* lrow = buf + (t * 16 + mcol) * ROW_STRIDE + half * 16;
      v8f acc = {};
#pragma unroll
      for (int kk = 0; kk < 8; ++kk) {
        v8bf b0 = *(const v8bf*)(lrow + kk * 64);
        v8bf b1 = *(const v8bf*)(lrow + kk * 64 + 32);
        v16bf B = __builtin_shufflevector(b0, b1, 0, 1, 2, 3, 4, 5, 6, 7,
                                          8, 9, 10, 11, 12, 13, 14, 15);
        acc = __builtin_amdgcn_wmma_f32_16x16x32_bf16(
            /*neg_a=*/false, A[kk], /*neg_b=*/false, B,
            /*c_mod=*/(short)0, acc, /*reuse_a=*/false, /*reuse_b=*/false);
      }
      const int c = ct * 16 + mcol;
      const float neg = e2[c];
#pragma unroll
      for (int v = 0; v < 8; ++v) {
        float sc = 2.0f * acc[v] - neg;              // monotone in -dist^2
        if (sc > best[v]) { best[v] = sc; bidx[v] = c; }  // '>' keeps earliest
      }
    }
    __syncthreads();   // everyone done reading buf before it is re-staged
  }

  // ---- cross-lane argmax; reduction scratch aliases the tile buffers ----
  // C/D layout: lane = column n, vgpr v = row m (+8 for lanes 16-31).
  float* s_score = (float*)smem;                 // [wave][v][lane] : 16 KB
  int*   s_idx   = (int*)(smem + 16384);         // [wave][v][lane] : 16 KB
  int*   s_code  = (int*)(smem + 32768);         // [wave][16]      : 1 KB
#pragma unroll
  for (int v = 0; v < 8; ++v) {
    s_score[(wave * 8 + v) * 32 + lane] = best[v];
    s_idx[(wave * 8 + v) * 32 + lane]   = bidx[v];
  }
  __syncthreads();

  if (lane < 16) {
    const int v  = lane & 7;
    const int lb = (lane < 8) ? 0 : 16;
    float bs = -3.4e38f;
    int   bi = 0x7fffffff;
#pragma unroll
    for (int j = 0; j < 16; ++j) {
      float sc = s_score[(wave * 8 + v) * 32 + lb + j];
      int   id = s_idx[(wave * 8 + v) * 32 + lb + j];
      if (sc > bs || (sc == bs && id < bi)) { bs = sc; bi = id; }
    }
    ind_out[row0 + lane] = (float)bi;   // tuple output #2 (indices, as float)
    s_code[wave * 16 + lane] = bi;
  }
  __syncthreads();

  // ---- gather: quantize[row,:] = embed_fp32[code,:], 8 floats per lane ----
  for (int r = 0; r < 16; ++r) {
    const int c = s_code[wave * 16 + r];
    const float* src = embed + (size_t)c * D_DIM + lane * 8;
    float* dst = quant + (size_t)(row0 + r) * D_DIM + lane * 8;
    *(v4f*)dst       = *(const v4f*)src;
    *(v4f*)(dst + 4) = *(const v4f*)(src + 4);
  }
}

// ---------------------------------------------------------------------------
extern "C" void kernel_launch(void* const* d_in, const int* in_sizes, int n_in,
                              void* d_out, int out_size, void* d_ws, size_t ws_size,
                              hipStream_t stream) {
  (void)n_in; (void)out_size; (void)ws_size;
  const float* x     = (const float*)d_in[0];   // [16,4096,256] fp32
  const float* embed = (const float*)d_in[1];   // [1,2048,256]  fp32
  const int D  = D_DIM;
  const int BN = in_sizes[0] / D;               // 65536
  const int C  = in_sizes[1] / D;               // 2048

  // workspace: [embed_bf16 : C*D*2 bytes][e2 : C*4 bytes]
  __bf16* ebf = (__bf16*)d_ws;
  float*  e2  = (float*)((char*)d_ws + (size_t)C * D * sizeof(unsigned short));

  float* qout = (float*)d_out;                  // quantize [16,4096,256]
  float* iout = qout + (size_t)BN * D;          // embed_ind [16,4096] as float

  vq_prep_kernel<<<C / 8, 256, 0, stream>>>(embed, ebf, e2);
  vq_argmin_kernel<<<BN / (16 * NWAVES), NTHREADS, 0, stream>>>(x, embed, ebf, e2,
                                                                qout, iout, C);
}
